// UNet_1305670058675
// MI455X (gfx1250) — matrix operand order repaired and checked by reading, not compiled, not run on hardware
//
#include <hip/hip_runtime.h>
#include <hip/hip_bf16.h>
#include <math.h>

#define B_SZ  32
#define L_SEQ 6400
#define T_TOK (B_SZ * L_SEQ)   // 204800 tokens
#define CH    64
#define NCHUNK 64
#define CS    (L_SEQ / NCHUNK) // 100 steps per chunk

typedef __bf16 bf16;
typedef __attribute__((ext_vector_type(16))) __bf16 v16bf;
typedef __attribute__((ext_vector_type(8)))  __bf16 v8bf;
typedef __attribute__((ext_vector_type(8)))  float  v8f;

__device__ __forceinline__ float siluf(float v)     { return v / (1.f + expf(-v)); }
__device__ __forceinline__ float softplusf(float v) { return (v > 20.f) ? v : log1pf(expf(v)); }

// ---------------- weight f32 -> bf16 ----------------
__global__ void k_f32_to_bf16(const float* __restrict__ s, bf16* __restrict__ d, int n) {
  int i = blockIdx.x * blockDim.x + threadIdx.x;
  if (i < n) d[i] = (bf16)s[i];
}

// ---------------- (B,C,HW) -> (B*HW, C) bf16 ----------------
__global__ void k_build_xs(const float* __restrict__ x, bf16* __restrict__ xs) {
  int idx = blockIdx.x * blockDim.x + threadIdx.x;      // coalesced over x
  if (idx >= B_SZ * CH * L_SEQ) return;
  int b   = idx / (CH * L_SEQ);
  int rem = idx - b * (CH * L_SEQ);
  int c   = rem / L_SEQ;
  int s   = rem - c * L_SEQ;
  xs[((size_t)(b * L_SEQ + s)) * CH + c] = (bf16)x[idx];
}

// ---------------- WMMA GEMM: C(T x NT*16) = A(T x 64) * W(NT*16 x 64)^T ----------------
// One wave per 16-row strip: A fragments loaded ONCE into registers, loop over all
// N-tiles (weights are L2/WGP$-resident).  K=64 -> 2x v_wmma_f32_16x16x32_bf16 per tile.
enum { EPI_F32 = 0, EPI_SILU_F32 = 1, EPI_SILU_BF16 = 2 };

template <int NT, int EPI>
__global__ __launch_bounds__(256)
void k_gemm_bf16(const bf16* __restrict__ A, const bf16* __restrict__ W,
                 float* __restrict__ Cf, bf16* __restrict__ Cb) {
  const int ldn = NT * 16;
  int wave = blockIdx.x * 8 + (threadIdx.x >> 5);   // strip index (grid sized exactly)
  int lane = threadIdx.x & 31;
  if (wave >= T_TOK / 16) return;                   // wave-uniform
  int ln   = lane & 15;
  int half = lane >> 4;

  // A fragment (16-bit A 16x32 layout): lane(ln,half) holds row m0+ln,
  //   K = 32*kb + 8*half + [0..7]  and  K = 32*kb + 16 + 8*half + [0..7]
  const bf16* arow = A + ((size_t)(wave * 16 + ln)) * 64 + 8 * half;
  v16bf a[2];
#pragma unroll
  for (int kb = 0; kb < 2; ++kb) {
    v8bf lo = *(const v8bf*)(arow + 32 * kb);
    v8bf hi = *(const v8bf*)(arow + 32 * kb + 16);
#pragma unroll
    for (int i = 0; i < 8; ++i) { a[kb][i] = lo[i]; a[kb][8 + i] = hi[i]; }
  }

  for (int ntile = 0; ntile < NT; ++ntile) {
    // B fragment (16-bit B 32x16 layout): lane(ln,half) holds col n0+ln,
    //   K = 32*kb + 8*half + [0..7]  and  K = 32*kb + 16 + 8*half + [0..7]  (W row-major (N,64))
    const bf16* brow = W + ((size_t)(ntile * 16 + ln)) * 64 + 8 * half;
    v8f acc = {};
#pragma unroll
    for (int kb = 0; kb < 2; ++kb) {
      v8bf blo = *(const v8bf*)(brow + 32 * kb);
      v8bf bhi = *(const v8bf*)(brow + 32 * kb + 16);
      v16bf b;
#pragma unroll
      for (int i = 0; i < 8; ++i) { b[i] = blo[i]; b[8 + i] = bhi[i]; }
      acc = __builtin_amdgcn_wmma_f32_16x16x32_bf16(false, a[kb], false, b, (short)0, acc, false, false);
    }
    // D layout: VGPR r -> row m0 + r + 8*half, col n0 + ln
    int col = ntile * 16 + ln;
#pragma unroll
    for (int r = 0; r < 8; ++r) {
      size_t off = ((size_t)(wave * 16 + r + 8 * half)) * ldn + col;
      float v = acc[r];
      if (EPI == EPI_SILU_F32 || EPI == EPI_SILU_BF16) v = siluf(v);
      if (EPI == EPI_SILU_BF16) Cb[off] = (bf16)v;
      else                      Cf[off] = v;
    }
  }
}

// ---------------- depthwise causal conv k=4 + bias + silu ----------------
__global__ void k_conv_silu(const float* __restrict__ xz, const float* __restrict__ cw,
                            const float* __restrict__ cb, float* __restrict__ out) {
  int idx = blockIdx.x * blockDim.x + threadIdx.x;
  if (idx >= T_TOK * CH) return;
  int c   = idx & 63;
  int tok = idx >> 6;
  int s   = tok % L_SEQ;
  float acc = cb[c];
#pragma unroll
  for (int k = 0; k < 4; ++k) {
    int ss = s - 3 + k;                       // causal left-pad 3, stays inside batch
    if (ss >= 0) acc += cw[c * 4 + k] * xz[(size_t)(tok - 3 + k) * 128 + c];
  }
  out[idx] = siluf(acc);
}

// ---------------- dbc = xc @ x_w^T (8), dt = softplus(dbc[:4] @ dt_w^T + dt_b) ----------------
__global__ void k_small_proj(const float* __restrict__ xcp, const float* __restrict__ x_w,
                             const float* __restrict__ dt_w, const float* __restrict__ dt_b,
                             float* __restrict__ dt_out, float* __restrict__ bc_out) {
  int t = blockIdx.x * blockDim.x + threadIdx.x;
  if (t >= T_TOK) return;
  const float* xr = xcp + (size_t)t * CH;
  float dbc[8] = {0.f, 0.f, 0.f, 0.f, 0.f, 0.f, 0.f, 0.f};
  for (int c = 0; c < CH; ++c) {
    float xv = xr[c];
#pragma unroll
    for (int j = 0; j < 8; ++j) dbc[j] += xv * x_w[j * CH + c];
  }
#pragma unroll
  for (int n = 0; n < 4; ++n) bc_out[(size_t)t * 4 + n] = dbc[4 + n];  // B0,B1,C0,C1
  for (int d = 0; d < CH; ++d) {
    float s = dt_b[d];
#pragma unroll
    for (int r = 0; r < 4; ++r) s += dbc[r] * dt_w[d * 4 + r];
    dt_out[(size_t)t * CH + d] = softplusf(s);
  }
}

// ---------------- chunked parallel selective scan (d_state=2) ----------------
// h_t = exp(dt_t*A)*h_{t-1} + dt_t*B_t*x_t.  Chunk transfer operator is
// (P, Q) with P = exp(A * sum(dt)) (product of exps collapses to exp-of-sum).
// Phase 1: per-chunk (S=sum dt, Q=h with h0=0).  4096 waves in flight.
__global__ __launch_bounds__(64)
void k_scan_phase1(const float* __restrict__ dt, const float* __restrict__ xcp,
                   const float* __restrict__ bc, const float* __restrict__ A_log,
                   float* __restrict__ Sarr, float* __restrict__ Qarr) {
  int blk = blockIdx.x;                       // b*NCHUNK + chunk
  int b = blk / NCHUNK, ck = blk - b * NCHUNK;
  int d = threadIdx.x;
  float A0 = -expf(A_log[d * 2 + 0]);
  float A1 = -expf(A_log[d * 2 + 1]);
  float S = 0.f, Q0 = 0.f, Q1 = 0.f;
  size_t base = (size_t)b * L_SEQ + (size_t)ck * CS;
  for (int t = 0; t < CS; ++t) {
    size_t tok = base + t;
    float dtv = dt[tok * CH + d];
    float xv  = xcp[tok * CH + d];
    float B0  = bc[tok * 4 + 0], B1 = bc[tok * 4 + 1];
    Q0 = expf(dtv * A0) * Q0 + dtv * B0 * xv;
    Q1 = expf(dtv * A1) * Q1 + dtv * B1 * xv;
    S += dtv;
  }
  size_t o = (size_t)blk * CH + d;
  Sarr[o] = S;
  Qarr[o * 2 + 0] = Q0;
  Qarr[o * 2 + 1] = Q1;
}

// Phase 2: sequential combine over 64 chunks per (b,d) -> chunk-initial states.
__global__ __launch_bounds__(64)
void k_scan_phase2(const float* __restrict__ Sarr, const float* __restrict__ Qarr,
                   const float* __restrict__ A_log, float* __restrict__ Hinit) {
  int b = blockIdx.x;
  int d = threadIdx.x;
  float A0 = -expf(A_log[d * 2 + 0]);
  float A1 = -expf(A_log[d * 2 + 1]);
  float h0 = 0.f, h1 = 0.f;
  for (int ck = 0; ck < NCHUNK; ++ck) {
    size_t o = (size_t)(b * NCHUNK + ck) * CH + d;
    Hinit[o * 2 + 0] = h0;
    Hinit[o * 2 + 1] = h1;
    float S = Sarr[o];
    h0 = expf(A0 * S) * h0 + Qarr[o * 2 + 0];
    h1 = expf(A1 * S) * h1 + Qarr[o * 2 + 1];
  }
}

// Phase 3: replay chunk from Hinit, emit y = (h.C + D*x) * silu(z) as bf16.
__global__ __launch_bounds__(64)
void k_scan_phase3(const float* __restrict__ dt, const float* __restrict__ xcp,
                   const float* __restrict__ bc, const float* __restrict__ xz,
                   const float* __restrict__ A_log, const float* __restrict__ Dp,
                   const float* __restrict__ Hinit, bf16* __restrict__ ybf) {
  int blk = blockIdx.x;
  int b = blk / NCHUNK, ck = blk - b * NCHUNK;
  int d = threadIdx.x;
  float A0 = -expf(A_log[d * 2 + 0]);
  float A1 = -expf(A_log[d * 2 + 1]);
  float Dd = Dp[d];
  size_t o = (size_t)blk * CH + d;
  float h0 = Hinit[o * 2 + 0], h1 = Hinit[o * 2 + 1];
  size_t base = (size_t)b * L_SEQ + (size_t)ck * CS;
  for (int t = 0; t < CS; ++t) {
    size_t tok = base + t;
    float dtv = dt[tok * CH + d];
    float xv  = xcp[tok * CH + d];
    float B0  = bc[tok * 4 + 0], B1 = bc[tok * 4 + 1];
    float C0  = bc[tok * 4 + 2], C1 = bc[tok * 4 + 3];
    h0 = expf(dtv * A0) * h0 + dtv * B0 * xv;
    h1 = expf(dtv * A1) * h1 + dtv * B1 * xv;
    float y = h0 * C0 + h1 * C1 + Dd * xv;
    float zv = xz[tok * 128 + 64 + d];
    ybf[tok * CH + d] = (bf16)(y * siluf(zv));
  }
}

// ---------------- layernorm: one wave32 per token, shuffle reductions ----------------
template <int BF16OUT>
__global__ __launch_bounds__(256)
void k_layernorm(const float* __restrict__ X, const float* __restrict__ g,
                 const float* __restrict__ bta, float* __restrict__ Of,
                 bf16* __restrict__ Ob) {
  int wave = blockIdx.x * 8 + (threadIdx.x >> 5);
  int lane = threadIdx.x & 31;
  if (wave >= T_TOK) return;
  const float* xr = X + (size_t)wave * CH;
  float2 v = *(const float2*)(xr + lane * 2);          // coalesced 256B per wave
  float s = v.x + v.y;
#pragma unroll
  for (int o = 16; o > 0; o >>= 1) s += __shfl_xor(s, o, 32);
  float mu = s * (1.f / CH);
  float dx = v.x - mu, dy = v.y - mu;
  float q = dx * dx + dy * dy;
#pragma unroll
  for (int o = 16; o > 0; o >>= 1) q += __shfl_xor(q, o, 32);
  float inv = rsqrtf(q * (1.f / CH) + 1e-5f);
  float o0 = dx * inv * g[lane * 2 + 0] + bta[lane * 2 + 0];
  float o1 = dy * inv * g[lane * 2 + 1] + bta[lane * 2 + 1];
  size_t off = (size_t)wave * CH + lane * 2;
  if (BF16OUT) { Ob[off] = (bf16)o0; Ob[off + 1] = (bf16)o1; }
  else         { Of[off] = o0;       Of[off + 1] = o1; }
}

// ---------------- out = silu(res + t), (T,C) -> (B,C,HW) ----------------
__global__ void k_combine(const float* __restrict__ res, const float* __restrict__ t2,
                          float* __restrict__ out) {
  int idx = blockIdx.x * blockDim.x + threadIdx.x;    // coalesced over out (B,C,HW)
  if (idx >= B_SZ * CH * L_SEQ) return;
  int b   = idx / (CH * L_SEQ);
  int rem = idx - b * (CH * L_SEQ);
  int c   = rem / L_SEQ;
  int s   = rem - c * L_SEQ;
  size_t toff = ((size_t)(b * L_SEQ + s)) * CH + c;
  out[idx] = siluf(res[toff] + t2[toff]);
}

// =====================================================================

static inline int cdiv(long a, int b) { return (int)((a + b - 1) / b); }

static void mamba_pass(const bf16* in_bf, const bf16* inw_b,
                       const float* conv_w, const float* conv_b,
                       const float* x_w, const float* dt_w, const float* dt_b,
                       const float* A_log, const float* Dp, const bf16* outw_b,
                       float* xz_f, float* xcp_f, float* dt_f, float* bc_f,
                       float* S_f, float* Q_f, float* H_f,
                       bf16* y_bf, float* t_f, hipStream_t stream) {
  const int strips = T_TOK / 16;                 // 12800 waves, 8 waves/block
  // xz = in @ in_w^T  (T x 128)
  k_gemm_bf16<8, EPI_F32><<<strips / 8, 256, 0, stream>>>(in_bf, inw_b, xz_f, nullptr);
  // xc = silu(causal depthwise conv(xz[:, :64]) + b)
  k_conv_silu<<<cdiv((long)T_TOK * CH, 256), 256, 0, stream>>>(xz_f, conv_w, conv_b, xcp_f);
  // dbc / softplus(dt)
  k_small_proj<<<cdiv(T_TOK, 256), 256, 0, stream>>>(xcp_f, x_w, dt_w, dt_b, dt_f, bc_f);
  // chunked scan
  k_scan_phase1<<<B_SZ * NCHUNK, 64, 0, stream>>>(dt_f, xcp_f, bc_f, A_log, S_f, Q_f);
  k_scan_phase2<<<B_SZ, 64, 0, stream>>>(S_f, Q_f, A_log, H_f);
  k_scan_phase3<<<B_SZ * NCHUNK, 64, 0, stream>>>(dt_f, xcp_f, bc_f, xz_f, A_log, Dp, H_f, y_bf);
  // t = y @ out_w^T  (T x 64)
  k_gemm_bf16<4, EPI_F32><<<strips / 8, 256, 0, stream>>>(y_bf, outw_b, t_f, nullptr);
}

extern "C" void kernel_launch(void* const* d_in, const int* in_sizes, int n_in,
                              void* d_out, int out_size, void* d_ws, size_t ws_size,
                              hipStream_t stream) {
  // setup_inputs() dict order (nested dicts flattened in insertion order):
  const float* x       = (const float*)d_in[0];
  const float* w_res   = (const float*)d_in[1];
  const float* w_mid   = (const float*)d_in[2];
  const float* in_w1   = (const float*)d_in[3];
  const float* conv_w1 = (const float*)d_in[4];
  const float* conv_b1 = (const float*)d_in[5];
  const float* x_w1    = (const float*)d_in[6];
  const float* dt_w1   = (const float*)d_in[7];
  const float* dt_b1   = (const float*)d_in[8];
  const float* A_log1  = (const float*)d_in[9];
  const float* D1      = (const float*)d_in[10];
  const float* out_w1  = (const float*)d_in[11];
  const float* in_w2   = (const float*)d_in[12];
  const float* conv_w2 = (const float*)d_in[13];
  const float* conv_b2 = (const float*)d_in[14];
  const float* x_w2    = (const float*)d_in[15];
  const float* dt_w2   = (const float*)d_in[16];
  const float* dt_b2   = (const float*)d_in[17];
  const float* A_log2  = (const float*)d_in[18];
  const float* D2      = (const float*)d_in[19];
  const float* out_w2  = (const float*)d_in[20];
  const float* ln1_g   = (const float*)d_in[21];
  const float* ln1_b   = (const float*)d_in[22];
  const float* ln2_g   = (const float*)d_in[23];
  const float* ln2_b   = (const float*)d_in[24];
  float* out = (float*)d_out;

  // ---- workspace layout (all regions 256B-aligned) ----
  char* ws = (char*)d_ws;
  bf16* w_res_b  = (bf16*)ws;             // 4096 elems
  bf16* w_mid_b  = w_res_b + 4096;
  bf16* in_w1_b  = w_res_b + 8192;        // 8192 elems
  bf16* out_w1_b = w_res_b + 16384;
  bf16* in_w2_b  = w_res_b + 20480;
  bf16* out_w2_b = w_res_b + 28672;       // -> 65536 bytes total
  size_t off = 65536;
  bf16*  xs_bf = (bf16*)(ws + off);  off += (size_t)T_TOK * CH * 2;   // reused for mid output
  float* res_f = (float*)(ws + off); off += (size_t)T_TOK * CH * 4;
  float* xz_f  = (float*)(ws + off); off += (size_t)T_TOK * 128 * 4;
  float* xcp_f = (float*)(ws + off); off += (size_t)T_TOK * CH * 4;
  float* dt_f  = (float*)(ws + off); off += (size_t)T_TOK * CH * 4;   // reused as ln2 f32 output
  float* bc_f  = (float*)(ws + off); off += (size_t)T_TOK * 4 * 4;
  bf16*  y_bf  = (bf16*)(ws + off);  off += (size_t)T_TOK * CH * 2;
  float* t_f   = (float*)(ws + off); off += (size_t)T_TOK * CH * 4;
  bf16*  ln_bf = (bf16*)(ws + off);  off += (size_t)T_TOK * CH * 2;
  float* S_f   = (float*)(ws + off); off += (size_t)B_SZ * NCHUNK * CH * 4;
  float* Q_f   = (float*)(ws + off); off += (size_t)B_SZ * NCHUNK * CH * 8;
  float* H_f   = (float*)(ws + off); off += (size_t)B_SZ * NCHUNK * CH * 8;
  (void)ws_size; (void)in_sizes; (void)n_in; (void)out_size;

  // ---- weights to bf16 ----
  k_f32_to_bf16<<<cdiv(4096, 256), 256, 0, stream>>>(w_res,  w_res_b,  4096);
  k_f32_to_bf16<<<cdiv(4096, 256), 256, 0, stream>>>(w_mid,  w_mid_b,  4096);
  k_f32_to_bf16<<<cdiv(8192, 256), 256, 0, stream>>>(in_w1,  in_w1_b,  8192);
  k_f32_to_bf16<<<cdiv(4096, 256), 256, 0, stream>>>(out_w1, out_w1_b, 4096);
  k_f32_to_bf16<<<cdiv(8192, 256), 256, 0, stream>>>(in_w2,  in_w2_b,  8192);
  k_f32_to_bf16<<<cdiv(4096, 256), 256, 0, stream>>>(out_w2, out_w2_b, 4096);

  // ---- xs (T,64) bf16 ----
  k_build_xs<<<cdiv((long)B_SZ * CH * L_SEQ, 256), 256, 0, stream>>>(x, xs_bf);

  const int strips = T_TOK / 16;
  // ---- res = silu(xs @ w_res^T) ----
  k_gemm_bf16<4, EPI_SILU_F32><<<strips / 8, 256, 0, stream>>>(xs_bf, w_res_b, res_f, nullptr);

  // ---- mamba 1 ----
  mamba_pass(xs_bf, in_w1_b, conv_w1, conv_b1, x_w1, dt_w1, dt_b1, A_log1, D1, out_w1_b,
             xz_f, xcp_f, dt_f, bc_f, S_f, Q_f, H_f, y_bf, t_f, stream);

  // ---- ln1 -> bf16 ----
  k_layernorm<1><<<cdiv(T_TOK, 8), 256, 0, stream>>>(t_f, ln1_g, ln1_b, nullptr, ln_bf);

  // ---- mid = silu(ln1 @ w_mid^T) -> bf16 (reuse xs_bf) ----
  k_gemm_bf16<4, EPI_SILU_BF16><<<strips / 8, 256, 0, stream>>>(ln_bf, w_mid_b, nullptr, xs_bf);

  // ---- mamba 2 ----
  mamba_pass(xs_bf, in_w2_b, conv_w2, conv_b2, x_w2, dt_w2, dt_b2, A_log2, D2, out_w2_b,
             xz_f, xcp_f, dt_f, bc_f, S_f, Q_f, H_f, y_bf, t_f, stream);

  // ---- ln2 -> f32 (into dt_f, dead after scan2) ----
  k_layernorm<0><<<cdiv(T_TOK, 8), 256, 0, stream>>>(t_f, ln2_g, ln2_b, dt_f, nullptr);

  // ---- out = silu(res + ln2), back to (B,C,H,W) ----
  k_combine<<<cdiv((long)B_SZ * CH * L_SEQ, 256), 256, 0, stream>>>(res_f, dt_f, out);
}
